// Qwen3MoeAttention_76647986365147
// MI455X (gfx1250) — matrix operand order repaired and checked by reading, not compiled, not run on hardware
//
#include <hip/hip_runtime.h>

#define T_TOK 2048
#define HID   2048
#define NH    32
#define NKV   4
#define HD    128
#define QKVD  ((NH + 2 * NKV) * HD)   // 5120
#define OD    (NH * HD)               // 4096
#define GQ    (NH / NKV)              // 8
#define ATTN_SCALE 0.08838834764831845f  // 1/sqrt(128), folded into Q

typedef __attribute__((ext_vector_type(16))) __bf16 bf16x16;
typedef __attribute__((ext_vector_type(8)))  __bf16 bf16x8;
typedef __attribute__((ext_vector_type(8)))  float  f32x8;

static __device__ __forceinline__ bf16x16 cat8(bf16x8 lo, bf16x8 hi) {
  return __builtin_shufflevector(lo, hi, 0, 1, 2, 3, 4, 5, 6, 7,
                                 8, 9, 10, 11, 12, 13, 14, 15);
}
static __device__ __forceinline__ f32x8 wmma_bf16(bf16x16 a, bf16x16 b, f32x8 c) {
  return __builtin_amdgcn_wmma_f32_16x16x32_bf16(false, a, false, b,
                                                 (short)0, c, false, false);
}
// Async DMA: 16B global -> LDS per lane, tracked by ASYNCcnt (no VGPR data).
static __device__ __forceinline__ void async_ld_b128(const void* gptr,
                                                     void* lptr) {
  unsigned l = (unsigned)(uintptr_t)lptr;  // low 32 bits = LDS byte address
  asm volatile("global_load_async_to_lds_b128 %0, %1, off" ::"v"(l),
               "v"((unsigned long long)(uintptr_t)gptr)
               : "memory");
}

// ---------------------------------------------------------------------------
// NT GEMM: C[M,N] (f32) = A[M,K] @ B[N,K]^T, bf16 compute / f32 accumulate.
// A is f32 or bf16 (template), B is f32; converted to bf16 while staging LDS.
// Block tile 128x128x32, 256 threads = 8 waves (2x4), wave tile 64x32.
// Double-buffered LDS: one barrier per K step, global fetch for k+1 issued
// before the 8 WMMAs of step k.
// ---------------------------------------------------------------------------
template <bool A_IS_F32>
__global__ __launch_bounds__(256) void gemm_nt(const void* __restrict__ Ap,
                                               const float* __restrict__ B,
                                               float* __restrict__ C,
                                               int M, int N, int K) {
  constexpr int LDT = 40;  // LDS row stride in halfs (80B, 16B aligned)
  __shared__ __bf16 As[2][128 * LDT];
  __shared__ __bf16 Bs[2][128 * LDT];

  const int tid = threadIdx.x;
  const int lane = tid & 31;
  const int wave = __builtin_amdgcn_readfirstlane(tid >> 5);  // SGPR wave id
  const int wm = wave >> 2;  // 0..1
  const int wn = wave & 3;   // 0..3
  const int lh = lane & 15;
  const int hsel = lane >> 4;
  const int m0 = blockIdx.y * 128;
  const int n0 = blockIdx.x * 128;

  f32x8 acc[4][2] = {};
  float4 va[4];
  bf16x8 vah[2];
  float4 vb[4];

  auto fetch = [&](int k0) {
    if (A_IS_F32) {
      const float* A = (const float*)Ap;
#pragma unroll
      for (int i = 0; i < 4; i++) {
        int c = tid + 256 * i, row = c >> 3, seg = c & 7;
        va[i] = ((const float4*)(A + (size_t)(m0 + row) * K + k0))[seg];
      }
    } else {
      const __bf16* A = (const __bf16*)Ap;
#pragma unroll
      for (int i = 0; i < 2; i++) {
        int c = tid + 256 * i, row = c >> 2, seg = c & 3;
        vah[i] = ((const bf16x8*)(A + (size_t)(m0 + row) * K + k0))[seg];
      }
    }
#pragma unroll
    for (int i = 0; i < 4; i++) {
      int c = tid + 256 * i, row = c >> 3, seg = c & 7;
      vb[i] = ((const float4*)(B + (size_t)(n0 + row) * K + k0))[seg];
    }
  };
  auto stage = [&](int buf) {
    if (A_IS_F32) {
#pragma unroll
      for (int i = 0; i < 4; i++) {
        int c = tid + 256 * i, row = c >> 3, seg = c & 7;
        __bf16* d = &As[buf][row * LDT + seg * 4];
        d[0] = (__bf16)va[i].x; d[1] = (__bf16)va[i].y;
        d[2] = (__bf16)va[i].z; d[3] = (__bf16)va[i].w;
      }
    } else {
#pragma unroll
      for (int i = 0; i < 2; i++) {
        int c = tid + 256 * i, row = c >> 2, seg = c & 3;
        *(bf16x8*)&As[buf][row * LDT + seg * 8] = vah[i];
      }
    }
#pragma unroll
    for (int i = 0; i < 4; i++) {
      int c = tid + 256 * i, row = c >> 3, seg = c & 7;
      __bf16* d = &Bs[buf][row * LDT + seg * 4];
      d[0] = (__bf16)vb[i].x; d[1] = (__bf16)vb[i].y;
      d[2] = (__bf16)vb[i].z; d[3] = (__bf16)vb[i].w;
    }
  };

  fetch(0);
  stage(0);
  __syncthreads();

  const int nk = K / 32;
  for (int kb = 0; kb < nk; kb++) {
    if (kb + 1 < nk) fetch((kb + 1) * 32);  // overlap global with WMMAs
    const int buf = kb & 1;

    bf16x16 a[4];
#pragma unroll
    for (int tm = 0; tm < 4; tm++) {
      int row = wm * 64 + tm * 16 + lh;
      bf16x8 lo = *(const bf16x8*)&As[buf][row * LDT + hsel * 8];
      bf16x8 hi = *(const bf16x8*)&As[buf][row * LDT + 16 + hsel * 8];
      a[tm] = cat8(lo, hi);
    }
    bf16x16 b[2];
#pragma unroll
    for (int tn = 0; tn < 2; tn++) {
      int col = wn * 32 + tn * 16 + lh;
      bf16x8 lo = *(const bf16x8*)&Bs[buf][col * LDT + hsel * 8];
      bf16x8 hi = *(const bf16x8*)&Bs[buf][col * LDT + 16 + hsel * 8];
      b[tn] = cat8(lo, hi);
    }
#pragma unroll
    for (int tm = 0; tm < 4; tm++)
#pragma unroll
      for (int tn = 0; tn < 2; tn++)
        acc[tm][tn] = wmma_bf16(a[tm], b[tn], acc[tm][tn]);

    if (kb + 1 < nk) stage((kb + 1) & 1);  // writes other buffer: no race
    __syncthreads();
  }

#pragma unroll
  for (int tm = 0; tm < 4; tm++)
#pragma unroll
    for (int tn = 0; tn < 2; tn++)
#pragma unroll
      for (int r = 0; r < 8; r++) {
        int m = m0 + wm * 64 + tm * 16 + r + hsel * 8;
        int n = n0 + wn * 32 + tn * 16 + lh;
        C[(size_t)m * N + n] = acc[tm][tn][r];
      }
}

// ---------------------------------------------------------------------------
// Per-(token, head) RMSNorm + NeoX RoPE; writes bf16 Q (pre-scaled by
// 1/sqrt(HD)), K, and transposed V.  grid = (T, NH + 2*NKV), block = 128
// ---------------------------------------------------------------------------
__global__ __launch_bounds__(128) void qk_rope(const float* __restrict__ qkv,
                                               const int* __restrict__ positions,
                                               const float* __restrict__ qw,
                                               const float* __restrict__ kw,
                                               __bf16* __restrict__ Qo,
                                               __bf16* __restrict__ Ko,
                                               __bf16* __restrict__ Vt) {
  const int t = blockIdx.x;
  const int hh = blockIdx.y;  // uniform per block
  const int i = threadIdx.x;
  const float x = qkv[(size_t)t * QKVD + hh * HD + i];

  if (hh >= NH + NKV) {  // V head: store transposed [hkv][d][t]
    int hv = hh - NH - NKV;
    Vt[((size_t)hv * HD + i) * T_TOK + t] = (__bf16)x;
    return;
  }

  __shared__ float sh[HD];
  __shared__ float ssum[4];

  float v = x * x;
#pragma unroll
  for (int off = 16; off > 0; off >>= 1) v += __shfl_xor(v, off, 32);
  if ((i & 31) == 0) ssum[i >> 5] = v;
  __syncthreads();
  const float ss = ssum[0] + ssum[1] + ssum[2] + ssum[3];
  const float inv = rsqrtf(ss * (1.0f / (float)HD) + 1e-6f);
  const float w = (hh < NH) ? qw[i] : kw[i];
  sh[i] = x * inv * w;
  __syncthreads();

  const float pos = (float)positions[t];
  const int f = i & 63;
  const float ang = pos * __powf(1000000.0f, -(float)f * (1.0f / 64.0f));
  float sn, cs;
  __sincosf(ang, &sn, &cs);
  const float out = (i < 64) ? (sh[i] * cs - sh[i + 64] * sn)
                             : (sh[i] * cs + sh[i - 64] * sn);
  if (hh < NH)  // fold attention scale into Q (rotation is linear)
    Qo[((size_t)t * NH + hh) * HD + i] = (__bf16)(out * ATTN_SCALE);
  else
    Ko[((size_t)t * NKV + (hh - NH)) * HD + i] = (__bf16)out;
}

// ---------------------------------------------------------------------------
// Causal flash attention. grid = (NH, T/128), block = 256 (8 waves).
// K/V tiles (32 keys) staged block-cooperatively into double-buffered LDS via
// global_load_async_to_lds_b128 (ASYNCcnt); each wave owns one 16-row Q tile.
// S = Q K^T via 8 WMMAs, P·V via 8 WMMAs against transposed V from LDS.
// ---------------------------------------------------------------------------
__global__ __launch_bounds__(256) void flash_attn(const __bf16* __restrict__ Q,
                                                  const __bf16* __restrict__ Kc,
                                                  const __bf16* __restrict__ Vt,
                                                  __bf16* __restrict__ O) {
  const int h = blockIdx.x;
  const int hkv = h / GQ;
  const int tid = threadIdx.x;
  const int lane = tid & 31;
  const int wave = __builtin_amdgcn_readfirstlane(tid >> 5);  // SGPR wave id
  const int lh = lane & 15;
  const int hsel = lane >> 4;
  const int Bq = blockIdx.y * 128;
  const int q0 = Bq + wave * 16;  // scalar: causal guard is a SALU branch

  __shared__ __bf16 Kt[2][32 * HD];   // [key][d]   8KB per buffer
  __shared__ __bf16 Vtt[2][HD * 32];  // [d][kv]    8KB per buffer
  __shared__ __bf16 pstage[8][16 * 32];
  __bf16* ps = pstage[wave];

  const int S = (Bq + 128) / 32;  // 32-key stages covering the causal range

  // 4 async b128 per thread per stage: K tile (32x256B) + Vt tile (128x64B)
  auto issue = [&](int s) {
    const int j0 = s * 32;
    const int buf = s & 1;
#pragma unroll
    for (int i = 0; i < 2; i++) {
      int e = tid + 256 * i;
      int row = e >> 4, seg = e & 15;
      async_ld_b128(Kc + ((size_t)(j0 + row) * NKV + hkv) * HD + seg * 8,
                    &Kt[buf][row * HD + seg * 8]);
    }
#pragma unroll
    for (int i = 0; i < 2; i++) {
      int e = tid + 256 * i;
      int d = e >> 2, seg = e & 3;
      async_ld_b128(Vt + ((size_t)hkv * HD + d) * T_TOK + j0 + seg * 8,
                    &Vtt[buf][d * 32 + seg * 8]);
    }
  };

  issue(0);
  issue(1);  // S >= 4 always; DMA streams while Q lands in registers

  // resident Q fragments (16 rows x 128 d), A-layout
  bf16x16 qf[4];
  const __bf16* qrow = Q + ((size_t)(q0 + lh) * NH + h) * HD;
#pragma unroll
  for (int c = 0; c < 4; c++) {
    bf16x8 lo = *(const bf16x8*)(qrow + c * 32 + hsel * 8);
    bf16x8 hi = *(const bf16x8*)(qrow + c * 32 + 16 + hsel * 8);
    qf[c] = cat8(lo, hi);
  }

  f32x8 o[8] = {};
  float mrow[8], lrow[8];
#pragma unroll
  for (int r = 0; r < 8; r++) {
    mrow[r] = -__builtin_inff();
    lrow[r] = 0.0f;
  }

  for (int s = 0; s < S; s++) {
    const int j0 = s * 32;
    const int buf = s & 1;
    // drain exactly the current stage (4 async ops/stage, in-order per wave)
    if (s + 1 < S)
      asm volatile("s_wait_asynccnt 0x4" ::: "memory");
    else
      asm volatile("s_wait_asynccnt 0x0" ::: "memory");
    __syncthreads();

    if (j0 <= q0 + 15) {  // scalar causal guard; EXEC all-ones inside
      // ---- S = Q K^T: two independent accumulator chains ----
      f32x8 sc0 = {}, sc1 = {};
#pragma unroll
      for (int c = 0; c < 4; c++) {
        const __bf16* k0p = &Kt[buf][lh * HD + c * 32 + hsel * 8];
        const __bf16* k1p = &Kt[buf][(16 + lh) * HD + c * 32 + hsel * 8];
        bf16x16 kf0 = cat8(*(const bf16x8*)k0p, *(const bf16x8*)(k0p + 16));
        bf16x16 kf1 = cat8(*(const bf16x8*)k1p, *(const bf16x8*)(k1p + 16));
        sc0 = wmma_bf16(qf[c], kf0, sc0);
        sc1 = wmma_bf16(qf[c], kf1, sc1);
      }

      // ---- online softmax over the 32 new columns (scale pre-folded) ----
#pragma unroll
      for (int r = 0; r < 8; r++) {
        const int qi = q0 + r + hsel * 8;
        float v0 = (j0 + lh <= qi) ? sc0[r] : -__builtin_inff();
        float v1 = (j0 + 16 + lh <= qi) ? sc1[r] : -__builtin_inff();
        float mx = fmaxf(v0, v1);
#pragma unroll
        for (int off = 1; off < 16; off <<= 1)
          mx = fmaxf(mx, __shfl_xor(mx, off, 32));  // row lives in a half-wave
        const float nm = fmaxf(mrow[r], mx);
        const float p0 = __expf(v0 - nm);
        const float p1 = __expf(v1 - nm);
        ps[(r + hsel * 8) * 32 + lh] = (__bf16)p0;  // C-layout -> LDS
        ps[(r + hsel * 8) * 32 + 16 + lh] = (__bf16)p1;
        float sum = p0 + p1;
#pragma unroll
        for (int off = 1; off < 16; off <<= 1) sum += __shfl_xor(sum, off, 32);
        const float alpha = __expf(mrow[r] - nm);
        lrow[r] = lrow[r] * alpha + sum;
        mrow[r] = nm;
#pragma unroll
        for (int dt = 0; dt < 8; dt++) o[dt][r] *= alpha;
      }

      // wave-local LDS relayout fence (DS ops are in-order within a wave)
      asm volatile("s_wait_dscnt 0x0" ::: "memory");

      bf16x8 plo = *(const bf16x8*)(ps + lh * 32 + hsel * 8);
      bf16x8 phi = *(const bf16x8*)(ps + lh * 32 + 16 + hsel * 8);
      bf16x16 pf = cat8(plo, phi);

      // ---- O += P V : pairs of independent WMMA chains ----
#pragma unroll
      for (int dt = 0; dt < 8; dt += 2) {
        const __bf16* v0p = &Vtt[buf][(dt * 16 + lh) * 32 + hsel * 8];
        const __bf16* v1p = &Vtt[buf][((dt + 1) * 16 + lh) * 32 + hsel * 8];
        bf16x16 vf0 = cat8(*(const bf16x8*)v0p, *(const bf16x8*)(v0p + 16));
        bf16x16 vf1 = cat8(*(const bf16x8*)v1p, *(const bf16x8*)(v1p + 16));
        o[dt] = wmma_bf16(pf, vf0, o[dt]);
        o[dt + 1] = wmma_bf16(pf, vf1, o[dt + 1]);
      }
    }

    __syncthreads();                 // all waves done reading buf
    if (s + 2 < S) issue(s + 2);     // refill same buffer for stage s+2
  }

  float invl[8];
#pragma unroll
  for (int r = 0; r < 8; r++) invl[r] = 1.0f / lrow[r];
#pragma unroll
  for (int dt = 0; dt < 8; dt++)
#pragma unroll
    for (int r = 0; r < 8; r++)
      O[(size_t)(q0 + r + hsel * 8) * OD + h * HD + dt * 16 + lh] =
          (__bf16)(o[dt][r] * invl[r]);
}

// ---------------------------------------------------------------------------
extern "C" void kernel_launch(void* const* d_in, const int* in_sizes, int n_in,
                              void* d_out, int out_size, void* d_ws,
                              size_t ws_size, hipStream_t stream) {
  (void)in_sizes; (void)n_in; (void)out_size; (void)ws_size;
  const int* positions = (const int*)d_in[0];
  const float* hidden = (const float*)d_in[1];
  const float* w_qkv = (const float*)d_in[2];
  const float* w_o = (const float*)d_in[3];
  const float* qw = (const float*)d_in[4];
  const float* kw = (const float*)d_in[5];
  float* out = (float*)d_out;

  char* ws = (char*)d_ws;  // all chunks are 256B-aligned by construction
  float* qkv = (float*)ws;  ws += (size_t)T_TOK * QKVD * sizeof(float);   // 40 MB
  __bf16* Qb = (__bf16*)ws; ws += (size_t)T_TOK * NH * HD * 2;            // 16 MB
  __bf16* Kb = (__bf16*)ws; ws += (size_t)T_TOK * NKV * HD * 2;           //  2 MB
  __bf16* Vt = (__bf16*)ws; ws += (size_t)NKV * HD * T_TOK * 2;           //  2 MB
  __bf16* Ob = (__bf16*)ws; ws += (size_t)T_TOK * OD * 2;                 // 16 MB

  // 1) fused QKV projection: [T,HID] x [QKVD,HID]^T -> f32 [T,QKVD]
  gemm_nt<true><<<dim3(QKVD / 128, T_TOK / 128), 256, 0, stream>>>(
      hidden, w_qkv, qkv, T_TOK, QKVD, HID);

  // 2) per-head RMSNorm + RoPE, emit bf16 Q (pre-scaled), K, transposed V
  qk_rope<<<dim3(T_TOK, NH + 2 * NKV), 128, 0, stream>>>(
      qkv, positions, qw, kw, Qb, Kb, Vt);

  // 3) causal flash attention -> bf16 O [T, NH*HD]
  flash_attn<<<dim3(NH, T_TOK / 128), 256, 0, stream>>>(Qb, Kb, Vt, Ob);

  // 4) output projection: [T,OD] x [HID,OD]^T -> f32 d_out
  gemm_nt<false><<<dim3(HID / 128, T_TOK / 128), 256, 0, stream>>>(
      Ob, w_o, out, T_TOK, HID, OD);
}
